// uWuModel_20177756356871
// MI455X (gfx1250) — compile-verified, matrix operand order checked by reading, not run on hardware
//
#include <hip/hip_runtime.h>
#include <hip/hip_bf16.h>
#include <stdint.h>

// ---------------------------------------------------------------------------
// Model constants
// ---------------------------------------------------------------------------
#define NN 50000
#define EE 600000
#define DD 128
#define GG 256      // graphs
#define SLOPE 0.15f

typedef __attribute__((ext_vector_type(16))) _Float16 v16h;
typedef __attribute__((ext_vector_type(8)))  _Float16 v8h;
typedef __attribute__((ext_vector_type(8)))  float    v8f;

// ---------------------------------------------------------------------------
// WMMA GEMM:  out[M,Nout] = act( A[M,K](f16) @ Wt[Nout,K]^T (f16) + bias )
//   Wt stored transposed ([Nout,K] row-major) so the B operand tile load is
//   identical to the A tile load (both M=16 x K=32 per-lane layouts).
//   Block tile: BM=128 x BN=64, 8 waves (4x2), each wave 32x32 = 4 WMMAs/kstep.
//   All K values are multiples of 32 (edge_attr padded 7->32).
// ---------------------------------------------------------------------------
#define BM 128
#define BN 64
#define BK 32
#define LSTR 48   // LDS row stride in halves (96B: 16B-aligned, spreads banks)

__global__ __launch_bounds__(256)
void gemm_f16_wmma(const _Float16* __restrict__ A, int M, int K,
                   const _Float16* __restrict__ Wt, int Nout,
                   const float* __restrict__ bias, int act,
                   _Float16* __restrict__ outH, float* __restrict__ outF)
{
    __shared__ _Float16 As[BM * LSTR];
    __shared__ _Float16 Bs[BN * LSTR];
    const int tid  = threadIdx.x;
    const int lane = tid & 31;
    const int wave = tid >> 5;
    const int wm   = wave >> 1;        // 0..3
    const int wn   = wave & 1;         // 0..1
    const int m0   = blockIdx.x * BM;
    const int n0   = blockIdx.y * BN;
    const int half = lane >> 4;        // K-half selector (WMMA A layout)
    const int l16  = lane & 15;

    v8f acc[2][2] = {};

    for (int kb = 0; kb < K; kb += BK) {
        // --- stage A tile: 128 rows x 32 halves (512 x 16B chunks, 2/thread)
#pragma unroll
        for (int c = 0; c < 2; ++c) {
            int chunk = tid + c * 256;
            int row   = chunk >> 2;
            int coff  = (chunk & 3) * 8;
            v8h v = {};
            int gr = m0 + row;
            if (gr < M) v = *(const v8h*)(A + (size_t)gr * K + kb + coff);
            *(v8h*)(&As[row * LSTR + coff]) = v;
        }
        // --- stage Wt tile: 64 rows x 32 halves (256 x 16B chunks, 1/thread)
        {
            int row  = tid >> 2;
            int coff = (tid & 3) * 8;
            v8h v = {};
            int gn = n0 + row;
            if (gn < Nout) v = *(const v8h*)(Wt + (size_t)gn * K + kb + coff);
            *(v8h*)(&Bs[row * LSTR + coff]) = v;
        }
        __syncthreads();

        // --- per-wave operand fetch (WMMA 16x16x32 f16 per-lane layout)
        v16h a[2], b[2];
#pragma unroll
        for (int i = 0; i < 2; ++i) {
            int r = wm * 32 + i * 16 + l16;
            v8h lo = *(const v8h*)(&As[r * LSTR + half * 8]);
            v8h hi = *(const v8h*)(&As[r * LSTR + 16 + half * 8]);
#pragma unroll
            for (int t = 0; t < 8; ++t) { a[i][t] = lo[t]; a[i][8 + t] = hi[t]; }
        }
#pragma unroll
        for (int j = 0; j < 2; ++j) {
            int r = wn * 32 + j * 16 + l16;
            v8h lo = *(const v8h*)(&Bs[r * LSTR + half * 8]);
            v8h hi = *(const v8h*)(&Bs[r * LSTR + 16 + half * 8]);
#pragma unroll
            for (int t = 0; t < 8; ++t) { b[j][t] = lo[t]; b[j][8 + t] = hi[t]; }
        }
#pragma unroll
        for (int i = 0; i < 2; ++i)
#pragma unroll
            for (int j = 0; j < 2; ++j)
                acc[i][j] = __builtin_amdgcn_wmma_f32_16x16x32_f16(
                    false, a[i], false, b[j], (short)0, acc[i][j], false, false);
        __syncthreads();
    }

    // --- epilogue: C layout VGPR r -> M = 8*half + r ; N = l16
#pragma unroll
    for (int i = 0; i < 2; ++i) {
#pragma unroll
        for (int j = 0; j < 2; ++j) {
            int col = n0 + wn * 32 + j * 16 + l16;
            if (col >= Nout) continue;
            float bv = bias ? bias[col] : 0.0f;
#pragma unroll
            for (int r = 0; r < 8; ++r) {
                int row = m0 + wm * 32 + i * 16 + half * 8 + r;
                if (row >= M) continue;
                float v = acc[i][j][r] + bv;
                if (act == 1) v = v >= 0.f ? v : SLOPE * v;
                size_t o = (size_t)row * Nout + col;
                if (outH) outH[o] = (_Float16)v;
                if (outF) outF[o] = v;
            }
        }
    }
}

// ---------------------------------------------------------------------------
// Utility / elementwise kernels
// ---------------------------------------------------------------------------
__global__ void zero_f32_kernel(float* p, long long count) {
    long long i = (long long)blockIdx.x * blockDim.x + threadIdx.x;
    if (i < count) p[i] = 0.f;
}
__global__ void init_amax_kernel(int* p, int count) {
    int i = blockIdx.x * blockDim.x + threadIdx.x;
    if (i < count) p[i] = (int)0x80000000;
}
// pack W [K,Nout] f32 -> Wt [Nout,Kp] f16 (transpose + zero pad K->Kp)
__global__ void pack_wt_kernel(const float* __restrict__ w, _Float16* __restrict__ wt,
                               int K, int Nout, int Kp) {
    int i = blockIdx.x * blockDim.x + threadIdx.x;
    if (i >= Nout * Kp) return;
    int n = i / Kp, k = i - n * Kp;
    wt[i] = (_Float16)(k < K ? w[(size_t)k * Nout + n] : 0.f);
}
// edge_attr [E,7] f32 -> [E,32] f16 zero-padded
__global__ void pack_edge_kernel(const float* __restrict__ ea, _Float16* __restrict__ out,
                                 int Ecnt) {
    long long i = (long long)blockIdx.x * blockDim.x + threadIdx.x;
    if (i >= (long long)Ecnt * 32) return;
    int e = (int)(i >> 5), k = (int)(i & 31);
    out[i] = (_Float16)(k < 7 ? ea[(size_t)e * 7 + k] : 0.f);
}
// h = x * w + b   (K=1 projection)
__global__ void input_proj_kernel(const float* __restrict__ x, const float* __restrict__ w,
                                  const float* __restrict__ b,
                                  float* __restrict__ hF, _Float16* __restrict__ hH, int n) {
    long long i = (long long)blockIdx.x * blockDim.x + threadIdx.x;
    if (i >= (long long)n * DD) return;
    int nid = (int)(i >> 7), d = (int)(i & 127);
    float v = x[nid] * w[d] + b[d];
    hF[i] = v; hH[i] = (_Float16)v;
}

// ---------------------------------------------------------------------------
// LayerNorm (wave per row) with optional lrelu / residual / f32 output
// ---------------------------------------------------------------------------
__global__ __launch_bounds__(256)
void ln_act_kernel(const _Float16* __restrict__ in, int rows, int width,
                   const float* __restrict__ g, const float* __restrict__ b,
                   int lrelu, _Float16* __restrict__ outH,
                   const float* __restrict__ res, float* __restrict__ outF)
{
    int row  = (blockIdx.x * blockDim.x + threadIdx.x) >> 5;
    int lane = threadIdx.x & 31;
    if (row >= rows) return;
    int per = width >> 5;               // 4 (D) or 8 (2D)
    const _Float16* rp = in + (size_t)row * width;
    float vals[8];
    float s = 0.f, sq = 0.f;
    for (int i = 0; i < per; ++i) {
        float v = (float)rp[lane * per + i];
        vals[i] = v; s += v; sq += v * v;
    }
    for (int m = 1; m < 32; m <<= 1) { s += __shfl_xor(s, m, 32); sq += __shfl_xor(sq, m, 32); }
    float mean = s / (float)width;
    float var  = sq / (float)width - mean * mean;
    float rstd = rsqrtf(var + 1e-5f);
    for (int i = 0; i < per; ++i) {
        int c = lane * per + i;
        float v = (vals[i] - mean) * rstd * g[c] + b[c];
        if (lrelu) v = v >= 0.f ? v : SLOPE * v;
        size_t o = (size_t)row * width + c;
        if (res)  v += res[o];
        if (outF) outF[o] = v;
        if (outH) outH[o] = (_Float16)v;
    }
}

// ---------------------------------------------------------------------------
// BatchNorm over nodes (training stats): pass 1 sums, pass 2 apply(+lrelu,+res)
// ---------------------------------------------------------------------------
__global__ __launch_bounds__(256)
void bn_stats_kernel(const float* __restrict__ x, int n, float* __restrict__ sums)
{
    __shared__ float ls[256], lq[256];
    int d = threadIdx.x & 127;
    float s = 0.f, q = 0.f;
    for (int r = blockIdx.x * 2 + (threadIdx.x >> 7); r < n; r += gridDim.x * 2) {
        float v = x[(size_t)r * DD + d];
        s += v; q += v * v;
    }
    ls[threadIdx.x] = s; lq[threadIdx.x] = q;
    __syncthreads();
    if (threadIdx.x < 128) {
        atomicAdd(&sums[d],       ls[threadIdx.x] + ls[threadIdx.x + 128]);
        atomicAdd(&sums[128 + d], lq[threadIdx.x] + lq[threadIdx.x + 128]);
    }
}
__global__ void bn_apply_kernel(const float* __restrict__ x, int n,
                                const float* __restrict__ sums,
                                const float* __restrict__ g, const float* __restrict__ b,
                                const float* __restrict__ res,
                                float* __restrict__ outF, float* __restrict__ outF2,
                                _Float16* __restrict__ outH)
{
    long long i = (long long)blockIdx.x * blockDim.x + threadIdx.x;
    if (i >= (long long)n * DD) return;
    int d = (int)(i & 127);
    float m  = sums[d] / (float)n;
    float vv = sums[128 + d] / (float)n - m * m;
    float v = (x[i] - m) * rsqrtf(vv + 1e-5f) * g[d] + b[d];
    v = v >= 0.f ? v : SLOPE * v;
    if (res) v += res[i];
    outF[i] = v;
    if (outF2) outF2[i] = v;
    if (outH)  outH[i]  = (_Float16)v;
}

// ---------------------------------------------------------------------------
// GIN gather/gate/scatter and pre-MLP
// ---------------------------------------------------------------------------
__global__ void gin_gate_scatter(const _Float16* __restrict__ e, const float* __restrict__ h,
                                 const int* __restrict__ src, const int* __restrict__ dst,
                                 float* __restrict__ agg, int Ecnt)
{
    long long i = (long long)blockIdx.x * blockDim.x + threadIdx.x;
    if (i >= (long long)Ecnt * DD) return;
    int eid = (int)(i >> 7), d = (int)(i & 127);
    float ev = (float)e[i];
    float gate = 1.f / (1.f + __expf(-ev));
    float m = gate * h[(size_t)src[eid] * DD + d] + (1.f - gate) * ev;
    atomicAdd(&agg[(size_t)dst[eid] * DD + d], m);
}
__global__ void gin_premlp_kernel(const float* __restrict__ h, const float* __restrict__ agg,
                                  const float* __restrict__ epsp, _Float16* __restrict__ out, int n)
{
    long long i = (long long)blockIdx.x * blockDim.x + threadIdx.x;
    if (i >= (long long)n * DD) return;
    out[i] = (_Float16)((1.f + *epsp) * h[i] + agg[i]);
}

// ---------------------------------------------------------------------------
// Transformer attention pieces (H=2 heads, C=64)
// ---------------------------------------------------------------------------
__global__ __launch_bounds__(256)
void attn_logit_kernel(const float* __restrict__ q, const float* __restrict__ k,
                       const _Float16* __restrict__ ee,
                       const int* __restrict__ src, const int* __restrict__ dst,
                       float* __restrict__ logit, int* __restrict__ amax, int Ecnt)
{
    int eid  = (blockIdx.x * blockDim.x + threadIdx.x) >> 5;
    int lane = threadIdx.x & 31;
    if (eid >= Ecnt) return;
    int s = src[eid], t = dst[eid];
    float p = 0.f;
#pragma unroll
    for (int i = 0; i < 4; ++i) {
        int d = lane * 4 + i;                       // lanes 0-15: head0, 16-31: head1
        float key = k[(size_t)s * DD + d] + (float)ee[(size_t)eid * DD + d];
        p += q[(size_t)t * DD + d] * key;
    }
    for (int m = 1; m < 16; m <<= 1) p += __shfl_xor(p, m, 32);
    if ((lane & 15) == 0) {
        int hh = lane >> 4;
        float lg = p * 0.125f;                      // 1/sqrt(64)
        logit[(size_t)eid * 2 + hh] = lg;
        int f = __float_as_int(lg);
        int key = f ^ ((f >> 31) & 0x7fffffff);     // order-preserving int encoding
        atomicMax(&amax[t * 2 + hh], key);
    }
}
__global__ void attn_exp_kernel(float* __restrict__ logit, const int* __restrict__ amax,
                                const int* __restrict__ dst, float* __restrict__ denom, int Ecnt)
{
    long long i = (long long)blockIdx.x * blockDim.x + threadIdx.x;
    if (i >= (long long)Ecnt * 2) return;
    int eid = (int)(i >> 1), hh = (int)(i & 1);
    int t = dst[eid];
    int kk = amax[t * 2 + hh];
    float am = __int_as_float(kk ^ ((kk >> 31) & 0x7fffffff));
    float ex = __expf(logit[i] - am);
    logit[i] = ex;
    atomicAdd(&denom[t * 2 + hh], ex);
}
__global__ void attn_msg_kernel(const float* __restrict__ v, const _Float16* __restrict__ ee,
                                const float* __restrict__ ex, const float* __restrict__ denom,
                                const int* __restrict__ src, const int* __restrict__ dst,
                                float* __restrict__ agg, int Ecnt)
{
    long long i = (long long)blockIdx.x * blockDim.x + threadIdx.x;
    if (i >= (long long)Ecnt * DD) return;
    int eid = (int)(i >> 7), d = (int)(i & 127), hh = d >> 6;
    int s = src[eid], t = dst[eid];
    float alpha = ex[(size_t)eid * 2 + hh] / (denom[t * 2 + hh] + 1e-16f);
    float m = (v[(size_t)s * DD + d] + (float)ee[i]) * alpha;
    atomicAdd(&agg[(size_t)t * DD + d], m);
}
__global__ __launch_bounds__(256)
void attn_beta_kernel(const float* __restrict__ agg, const float* __restrict__ xr,
                      const float* __restrict__ bw, float* __restrict__ out, int n)
{
    int nid  = (blockIdx.x * blockDim.x + threadIdx.x) >> 5;
    int lane = threadIdx.x & 31;
    if (nid >= n) return;
    float p = 0.f, o4[4], x4[4];
#pragma unroll
    for (int i = 0; i < 4; ++i) {
        int d = lane * 4 + i;
        float o = agg[(size_t)nid * DD + d], xx = xr[(size_t)nid * DD + d];
        o4[i] = o; x4[i] = xx;
        p += o * bw[d] + xx * bw[128 + d] + (o - xx) * bw[256 + d];
    }
    for (int m = 1; m < 32; m <<= 1) p += __shfl_xor(p, m, 32);
    float beta = 1.f / (1.f + __expf(-p));
#pragma unroll
    for (int i = 0; i < 4; ++i) {
        int d = lane * 4 + i;
        out[(size_t)nid * DD + d] = beta * x4[i] + (1.f - beta) * o4[i];
    }
}

// ---------------------------------------------------------------------------
// Pooling + classifier
// ---------------------------------------------------------------------------
__global__ void pool_scatter_kernel(const float* __restrict__ h, const int* __restrict__ batch,
                                    float* __restrict__ pooled, float* __restrict__ cnt, int n)
{
    long long i = (long long)blockIdx.x * blockDim.x + threadIdx.x;
    if (i >= (long long)n * DD) return;
    int nid = (int)(i >> 7), d = (int)(i & 127);
    int g = batch[nid];
    atomicAdd(&pooled[(size_t)g * DD + d], h[i]);
    if (d == 0) atomicAdd(&cnt[g], 1.0f);
}
__global__ __launch_bounds__(64)
void cls_kernel(const float* __restrict__ pooled, const float* __restrict__ cnt,
                const float* __restrict__ w1, const float* __restrict__ b1,
                const float* __restrict__ w2, const float* __restrict__ b2,
                float* __restrict__ out)
{
    __shared__ float gbuf[128], hid[64];
    int gph = blockIdx.x, t = threadIdx.x;
    float inv = 1.f / fmaxf(cnt[gph], 1.f);
    gbuf[t]      = pooled[(size_t)gph * DD + t]      * inv;
    gbuf[t + 64] = pooled[(size_t)gph * DD + t + 64] * inv;
    __syncthreads();
    float s = b1[t];
    for (int k = 0; k < 128; ++k) s += gbuf[k] * w1[(size_t)k * 64 + t];
    hid[t] = s >= 0.f ? s : SLOPE * s;
    __syncthreads();
    if (t < 6) {
        float o = b2[t];
        for (int k = 0; k < 64; ++k) o += hid[k] * w2[(size_t)k * 6 + t];
        out[gph * 6 + t] = o;
    }
}

// ---------------------------------------------------------------------------
// Host orchestration
// ---------------------------------------------------------------------------
extern "C" void kernel_launch(void* const* d_in, const int* in_sizes, int n_in,
                              void* d_out, int out_size, void* d_ws, size_t ws_size,
                              hipStream_t stream)
{
    (void)in_sizes; (void)out_size; (void)ws_size;
    if (n_in < 98) return;

    // ---- input pointers (JAX pytree order: dict keys sorted alphabetically)
    const float* x          = (const float*)d_in[0];
    const int*   edge_index = (const int*)  d_in[1];
    const float* edge_attr  = (const float*)d_in[2];
    const int*   batch      = (const int*)  d_in[3];
    auto F = [&](int i) -> const float* { return (const float*)d_in[i]; };
    const int GIN0 = 16, GSTR = 17;
    auto GFp = [&](int L, int k) -> const float* { return (const float*)d_in[GIN0 + L * GSTR + k]; };
    // gin leaf offsets: 0 ee1.b 1 ee1.w 2 ee2.b 3 ee2.w 4 ee3.b 5 ee3.w
    //                   6 ee_ln.b 7 ee_ln.g 8 eps 9 mlp1.b 10 mlp1.w 11 mlp2.b
    //                   12 mlp2.w 13 mlp_ln1.b 14 mlp_ln1.g 15 mlp_ln2.b 16 mlp_ln2.g
    const int* src = edge_index;
    const int* dst = edge_index + EE;

    // ---- workspace layout
    size_t off = 0;
    auto alloc = [&](size_t bytes) -> void* {
        off = (off + 255) & ~(size_t)255;
        void* p = (char*)d_ws + off;
        off += bytes;
        return p;
    };
    _Float16* wt_ee1[4]; _Float16* wt_ee2[4]; _Float16* wt_ee3[4];
    _Float16* wt_mlp1[4]; _Float16* wt_mlp2[4];
    for (int L = 0; L < 4; ++L) {
        wt_ee1[L]  = (_Float16*)alloc(128 * 32  * 2);
        wt_ee2[L]  = (_Float16*)alloc(128 * 128 * 2);
        wt_ee3[L]  = (_Float16*)alloc(128 * 128 * 2);
        wt_mlp1[L] = (_Float16*)alloc(256 * 128 * 2);
        wt_mlp2[L] = (_Float16*)alloc(128 * 256 * 2);
    }
    _Float16* wt_te1  = (_Float16*)alloc(128 * 32  * 2);
    _Float16* wt_te2  = (_Float16*)alloc(128 * 128 * 2);
    _Float16* wt_q    = (_Float16*)alloc(128 * 128 * 2);
    _Float16* wt_k    = (_Float16*)alloc(128 * 128 * 2);
    _Float16* wt_v    = (_Float16*)alloc(128 * 128 * 2);
    _Float16* wt_e    = (_Float16*)alloc(128 * 128 * 2);
    _Float16* wt_skip = (_Float16*)alloc(128 * 128 * 2);

    _Float16* Ae  = (_Float16*)alloc((size_t)EE * 32  * 2);
    _Float16* eA  = (_Float16*)alloc((size_t)EE * DD * 2);
    _Float16* eB  = (_Float16*)alloc((size_t)EE * DD * 2);
    float* h_f32  = (float*)alloc((size_t)NN * DD * 4);
    float* res_f  = (float*)alloc((size_t)NN * DD * 4);
    float* agg_f  = (float*)alloc((size_t)NN * DD * 4);
    float* qf     = (float*)alloc((size_t)NN * DD * 4);
    float* kf     = (float*)alloc((size_t)NN * DD * 4);
    float* vf     = (float*)alloc((size_t)NN * DD * 4);
    float* xrf    = (float*)alloc((size_t)NN * DD * 4);
    float* tT     = (float*)alloc((size_t)NN * DD * 4);
    _Float16* nA  = (_Float16*)alloc((size_t)NN * DD * 2);
    _Float16* nB  = (_Float16*)alloc((size_t)NN * 256 * 2);
    _Float16* hH  = (_Float16*)alloc((size_t)NN * DD * 2);
    float* logit  = (float*)alloc((size_t)EE * 2 * 4);
    int*   amax   = (int*)  alloc((size_t)NN * 2 * 4);
    float* denom  = (float*)alloc((size_t)NN * 2 * 4);
    float* sums   = (float*)alloc(256 * 4);
    float* pooled = (float*)alloc((size_t)GG * DD * 4);
    float* cntg   = (float*)alloc(GG * 4);

    // ---- helpers
    auto zero = [&](float* p, long long count) {
        zero_f32_kernel<<<(unsigned)((count + 255) / 256), 256, 0, stream>>>(p, count);
    };
    auto gemm = [&](const _Float16* A, int M, int K, const _Float16* Wt, int Nn,
                    const float* bias, int act, _Float16* oH, float* oF) {
        dim3 g((M + BM - 1) / BM, (Nn + BN - 1) / BN);
        gemm_f16_wmma<<<g, 256, 0, stream>>>(A, M, K, Wt, Nn, bias, act, oH, oF);
    };
    auto pack = [&](const float* w, _Float16* wt, int K, int Nn, int Kp) {
        int total = Nn * Kp;
        pack_wt_kernel<<<(total + 255) / 256, 256, 0, stream>>>(w, wt, K, Nn, Kp);
    };
    auto ln = [&](const _Float16* in, int rows, int width, const float* g, const float* b,
                  int lrelu, _Float16* oH, const float* res, float* oF) {
        ln_act_kernel<<<(rows + 7) / 8, 256, 0, stream>>>(in, rows, width, g, b, lrelu, oH, res, oF);
    };
    auto batchnorm = [&](const float* xin, const float* g, const float* b, const float* res,
                         float* oF, float* oF2, _Float16* oH) {
        zero(sums, 256);
        bn_stats_kernel<<<512, 256, 0, stream>>>(xin, NN, sums);
        bn_apply_kernel<<<(NN * DD + 255) / 256, 256, 0, stream>>>(xin, NN, sums, g, b, res, oF, oF2, oH);
    };

    // ---- pack all GEMM weights to f16 (transposed, K padded to mult of 32)
    for (int L = 0; L < 4; ++L) {
        pack(GFp(L, 1),  wt_ee1[L],  7,   128, 32);
        pack(GFp(L, 3),  wt_ee2[L],  128, 128, 128);
        pack(GFp(L, 5),  wt_ee3[L],  128, 128, 128);
        pack(GFp(L, 10), wt_mlp1[L], 128, 256, 128);
        pack(GFp(L, 12), wt_mlp2[L], 256, 128, 256);
    }
    pack(F(93), wt_te1,  7,   128, 32);
    pack(F(95), wt_te2,  128, 128, 128);
    pack(F(89), wt_q,    128, 128, 128);
    pack(F(87), wt_k,    128, 128, 128);
    pack(F(97), wt_v,    128, 128, 128);
    pack(F(15), wt_e,    128, 128, 128);
    pack(F(91), wt_skip, 128, 128, 128);

    // ---- pack edge_attr [E,7] -> [E,32] f16
    pack_edge_kernel<<<(unsigned)(((long long)EE * 32 + 255) / 256), 256, 0, stream>>>(edge_attr, Ae, EE);

    // ---- input projection (K=1)
    input_proj_kernel<<<(NN * DD + 255) / 256, 256, 0, stream>>>(x, F(85), F(84), h_f32, hH, NN);

    // ---- GIN layer
    auto gin_layer = [&](int L) {
        gemm(Ae, EE, 32, wt_ee1[L], 128, GFp(L, 0), 0, eA, nullptr);
        ln(eA, EE, 128, GFp(L, 7), GFp(L, 6), 1, eA, nullptr, nullptr);          // LN+lrelu
        gemm(eA, EE, 128, wt_ee2[L], 128, GFp(L, 2), 1, eB, nullptr);            // lin+lrelu
        gemm(eB, EE, 128, wt_ee3[L], 128, GFp(L, 4), 0, eA, nullptr);            // e
        zero(agg_f, (long long)NN * DD);
        gin_gate_scatter<<<(unsigned)(((long long)EE * DD + 255) / 256), 256, 0, stream>>>(
            eA, h_f32, src, dst, agg_f, EE);
        gin_premlp_kernel<<<(NN * DD + 255) / 256, 256, 0, stream>>>(h_f32, agg_f, GFp(L, 8), nA, NN);
        gemm(nA, NN, 128, wt_mlp1[L], 256, GFp(L, 9), 0, nB, nullptr);
        ln(nB, NN, 256, GFp(L, 14), GFp(L, 13), 1, nB, nullptr, nullptr);
        gemm(nB, NN, 256, wt_mlp2[L], 128, GFp(L, 11), 0, nA, nullptr);
        ln(nA, NN, 128, GFp(L, 16), GFp(L, 15), 0, hH, h_f32, h_f32);            // h = x + LN(...)
    };

    gin_layer(0);
    gin_layer(1);

    // ---- bn1 + lrelu ; res = h
    batchnorm(h_f32, F(6), F(5), nullptr, h_f32, res_f, hH);

    // ---- transformer
    gemm(Ae, EE, 32, wt_te1, 128, F(92), 1, eA, nullptr);         // lrelu(lin(edge_attr))
    gemm(eA, EE, 128, wt_te2, 128, F(94), 0, eB, nullptr);        // e
    gemm(eB, EE, 128, wt_e,  128, nullptr, 0, eA, nullptr);       // ee (no bias)
    gemm(hH, NN, 128, wt_q,    128, F(88), 0, nullptr, qf);
    gemm(hH, NN, 128, wt_k,    128, F(86), 0, nullptr, kf);
    gemm(hH, NN, 128, wt_v,    128, F(96), 0, nullptr, vf);
    gemm(hH, NN, 128, wt_skip, 128, F(90), 0, nullptr, xrf);
    init_amax_kernel<<<(NN * 2 + 255) / 256, 256, 0, stream>>>(amax, NN * 2);
    attn_logit_kernel<<<(unsigned)(((long long)EE * 32 + 255) / 256), 256, 0, stream>>>(
        qf, kf, eA, src, dst, logit, amax, EE);
    zero(denom, (long long)NN * 2);
    attn_exp_kernel<<<(unsigned)(((long long)EE * 2 + 255) / 256), 256, 0, stream>>>(
        logit, amax, dst, denom, EE);
    zero(agg_f, (long long)NN * DD);
    attn_msg_kernel<<<(unsigned)(((long long)EE * DD + 255) / 256), 256, 0, stream>>>(
        vf, eA, logit, denom, src, dst, agg_f, EE);
    attn_beta_kernel<<<(NN * 32 + 255) / 256, 256, 0, stream>>>(agg_f, xrf, F(4), tT, NN);

    // ---- bnt + lrelu + residual
    batchnorm(tT, F(10), F(9), res_f, h_f32, nullptr, hH);

    gin_layer(2);
    gin_layer(3);

    // ---- bn2 + lrelu
    batchnorm(h_f32, F(8), F(7), nullptr, h_f32, nullptr, nullptr);

    // ---- global mean pool + classifier
    zero(pooled, (long long)GG * DD);
    zero(cntg, GG);
    pool_scatter_kernel<<<(NN * DD + 255) / 256, 256, 0, stream>>>(h_f32, batch, pooled, cntg, NN);
    cls_kernel<<<GG, 64, 0, stream>>>(pooled, cntg, F(12), F(11), F(14), F(13), (float*)d_out);
}
